// MaskedSpatialCrossAttention_17669495456331
// MI455X (gfx1250) — matrix operand order repaired and checked by reading, not compiled, not run on hardware
//
#include <hip/hip_runtime.h>

// ---------------------------------------------------------------------------
// Types for CDNA5 WMMA (wave32): 16x16x32 bf16 -> f32, plus TDM descriptors
// ---------------------------------------------------------------------------
typedef __attribute__((ext_vector_type(16))) __bf16       v16bf;
typedef __attribute__((ext_vector_type(8)))  __bf16       v8bf;
typedef __attribute__((ext_vector_type(8)))  float        v8f;
typedef __attribute__((ext_vector_type(4)))  unsigned int v4u;
typedef __attribute__((ext_vector_type(8)))  int          v8i;
typedef __attribute__((ext_vector_type(4)))  int          v4i;

#define B_   256
#define T_   384      // 2*LEN_T + S
#define S_   256
#define D_   768
#define LENT 64
#define MROWS (B_ * T_)   // 98304

// Staging-path selection (evaluated per compile pass; host pass falls through)
#if __has_builtin(__builtin_amdgcn_tensor_load_to_lds) && __has_builtin(__builtin_amdgcn_s_wait_tensorcnt)
#define STAGE_TDM 1
#elif __has_builtin(__builtin_amdgcn_global_load_async_to_lds_b128) && __has_builtin(__builtin_amdgcn_s_wait_asynccnt)
#define STAGE_ASYNC 1
#endif

// Padded LDS layout shared by all staging paths: 32 pad bytes every 256
// elements (matches TDM pad_interval=6 / pad_amount=7). Element index -> byte.
static __device__ __forceinline__ int lds_off(int l) {
    return 2 * l + 32 * (l >> 8);
}
#define BPANEL_ELEMS (64 * D_)                   // 49152 bf16
#define BPANEL_BYTES (2 * BPANEL_ELEMS + 32 * (BPANEL_ELEMS / 256))  // 104448
#define ROWPITCH 1632   // lds bytes per B row: 2*768 + 3 pads * 32

// round-to-nearest-even f32 -> bf16
static __device__ __forceinline__ __bf16 f2bf(float f) {
    union { float f; unsigned int u; } a; a.f = f;
    unsigned int r = a.u + 0x7FFFu + ((a.u >> 16) & 1u);
    union { unsigned short s; __bf16 b; } o; o.s = (unsigned short)(r >> 16);
    return o.b;
}

// block-wide reduction over 256 threads (op: 0=sum, 1=min, 2=max)
static __device__ __forceinline__ float block_op(float v, float* red, int tid, int op) {
    red[tid] = v; __syncthreads();
    #pragma unroll
    for (int s = 128; s > 0; s >>= 1) {
        if (tid < s) {
            float a = red[tid], b = red[tid + s];
            red[tid] = (op == 0) ? (a + b) : (op == 1 ? fminf(a, b) : fmaxf(a, b));
        }
        __syncthreads();
    }
    float r = red[0]; __syncthreads();
    return r;
}

// ---------------------------------------------------------------------------
// Phase 0: convert proj_W (768x768 f32) to bf16 (kept hot in L2, 1.2 MB)
// ---------------------------------------------------------------------------
__global__ __launch_bounds__(256) void k_cvt_bf16(const float* __restrict__ src,
                                                  __bf16* __restrict__ dst, int n) {
    int i = blockIdx.x * 256 + threadIdx.x;
    if (i < n) dst[i] = f2bf(src[i]);
}

// ---------------------------------------------------------------------------
// Phase 1: one block per batch -> W_proj[b, 256], w[b], wi[b]
// ---------------------------------------------------------------------------
__global__ __launch_bounds__(256) void k_stats(const float* __restrict__ x,
                                               const float* __restrict__ xi,
                                               const int*   __restrict__ mask,
                                               const float* __restrict__ pwW,
                                               const float* __restrict__ pwb,
                                               float* __restrict__ Wproj,
                                               float* __restrict__ wv,
                                               float* __restrict__ wiv) {
    __shared__ float q_rel[D_], q_irr[D_];
    __shared__ float T[4][S_];
    __shared__ float red[256];
    __shared__ float Wsh[S_];
    __shared__ int   smask[LENT];

    const int tid = threadIdx.x;
    const int b   = blockIdx.x;
    const size_t xb = (size_t)b * T_ * D_;

    if (tid < LENT) smask[tid] = mask[b * LENT + tid];
    __syncthreads();

    float mv = (tid < LENT) ? (float)smask[tid] : 0.f;
    float msum = block_op(mv, red, tid, 0);            // sum(mask[b]) over 64
    // cnt_rel = 2*msum (mask duplicated over 128 rows); q = s / (2*cnt)
    float inv_rel = 1.f / (4.f * msum);
    float inv_irr = 1.f / (2.f * (128.f - 2.f * msum));

    // masked means over template rows (t in [0,128)), x + xi combined
    for (int d = tid; d < D_; d += 256) {
        float sr = 0.f, si = 0.f;
        for (int t = 0; t < 2 * LENT; ++t) {
            float v = x[xb + (size_t)t * D_ + d] + xi[xb + (size_t)t * D_ + d];
            if (smask[t & (LENT - 1)]) sr += v; else si += v;
        }
        q_rel[d] = sr * inv_rel;
        q_irr[d] = si * inv_irr;
    }
    __syncthreads();

    // attention rows: one wave per search row; exp(softplus(z)) = 1 + e^z
    const int wave = tid >> 5, lane = tid & 31;
    const float scale = rsqrtf((float)D_);
    for (int s = wave; s < S_; s += 8) {
        float dr = 0.f, dri = 0.f, di = 0.f, dii = 0.f;
        const float* sx  = &x [xb + (size_t)(2 * LENT + s) * D_];
        const float* sxi = &xi[xb + (size_t)(2 * LENT + s) * D_];
        for (int d = lane; d < D_; d += 32) {
            float qr = q_rel[d], qi = q_irr[d];
            float a = sx[d], c = sxi[d];
            dr += qr * a; dri += qr * c; di += qi * a; dii += qi * c;
        }
        #pragma unroll
        for (int off = 16; off > 0; off >>= 1) {
            dr  += __shfl_down(dr,  off, 32);
            dri += __shfl_down(dri, off, 32);
            di  += __shfl_down(di,  off, 32);
            dii += __shfl_down(dii, off, 32);
        }
        if (lane == 0) {
            T[0][s] = 1.f + expf(dr  * scale);
            T[1][s] = 1.f + expf(dri * scale);
            T[2][s] = 1.f + expf(di  * scale);
            T[3][s] = 1.f + expf(dii * scale);
        }
    }
    __syncthreads();

    float t0 = T[0][tid], t1 = T[1][tid], t2 = T[2][tid], t3 = T[3][tid];
    float mn, mx;
    mn = block_op(t0, red, tid, 1); mx = block_op(t0, red, tid, 2);
    float TRs  = (t0 - mn) / (mx - mn) * 2.f;
    mn = block_op(t1, red, tid, 1); mx = block_op(t1, red, tid, 2);
    float TRis = (t1 - mn) / (mx - mn) * 2.f;
    mn = block_op(t2, red, tid, 1); mx = block_op(t2, red, tid, 2);
    float TIs  = (t2 - mn) / (mx - mn);
    mn = block_op(t3, red, tid, 1); mx = block_op(t3, red, tid, 2);
    float TIis = (t3 - mn) / (mx - mn);

    float meanTR  = block_op(TRs,  red, tid, 0) * (1.f / S_);
    float meanTRi = block_op(TRis, red, tid, 0) * (1.f / S_);
    float denom = meanTR + meanTRi;
    float w  = meanTRi / denom;
    float wi = meanTR  / denom;

    float Wv = w * (TRs - TIs) + wi * (TRis - TIis);
    Wsh[tid] = Wv; __syncthreads();

    float o = pwb[tid];
    for (int s = 0; s < S_; ++s) o += Wsh[s] * pwW[(size_t)tid * S_ + s];
    mn = block_op(o, red, tid, 1); mx = block_op(o, red, tid, 2);
    Wproj[b * S_ + tid] = (o - mn) / (mx - mn) * 2.f;

    if (tid == 0) { wv[b] = w; wiv[b] = wi; }
}

// ---------------------------------------------------------------------------
// Phase 2: x_out row coefficients (g, gi) + bf16 A-matrix for the WMMA GEMM
// ---------------------------------------------------------------------------
__global__ __launch_bounds__(256) void k_build_a(const float* __restrict__ x,
                                                 const float* __restrict__ xi,
                                                 const int*   __restrict__ mask,
                                                 const float* __restrict__ Wproj,
                                                 const float* __restrict__ wv,
                                                 const float* __restrict__ wiv,
                                                 __bf16* __restrict__ Ab,
                                                 float* __restrict__ g,
                                                 float* __restrict__ gi) {
    const int row = blockIdx.x;            // b*384 + t
    const int b = row / T_, t = row % T_;
    const float w = wv[b], wi = wiv[b];
    float gg, ggi;
    if (t < 2 * LENT) {
        float gain = 1.f + (float)mask[b * LENT + (t & (LENT - 1))];
        gg = gain * w; ggi = gain * wi;
    } else {
        float wp = Wproj[b * S_ + (t - 2 * LENT)];
        gg = wp * w; ggi = wp * wi;
    }
    if (threadIdx.x == 0) { g[row] = gg; gi[row] = ggi; }
    const size_t ro = (size_t)row * D_;
    for (int d = threadIdx.x; d < D_; d += 256) {
        Ab[ro + d] = f2bf(gg * x[ro + d] + ggi * xi[ro + d]);
    }
}

// ---------------------------------------------------------------------------
// Phase 3: out = x_out + x_out @ proj_W.T + proj_b   via v_wmma_f32_16x16x32_bf16
//   grid (N/64, M/128); 8 waves/block, wave tile = 16x64 (4 accumulators).
//   proj_W panel (64 contiguous rows of Wb) staged into padded LDS via the
//   Tensor Data Mover (TENSORcnt), async-to-LDS (ASYNCcnt), or plain copy.
//   B-fragment LDS addresses: loop-invariant per-lane base + compile-time
//   constant per-kt offset, so ds_load_b128 gets immediate offsets and the
//   K loop fully unrolls (96 WMMAs, no per-iteration VALU address math).
// ---------------------------------------------------------------------------
__global__ __launch_bounds__(256) void k_gemm(const __bf16* __restrict__ Ab,
                                              __bf16* __restrict__ Wb,
                                              const float* __restrict__ x,
                                              const float* __restrict__ xi,
                                              const float* __restrict__ g,
                                              const float* __restrict__ gi,
                                              const float* __restrict__ projb,
                                              float* __restrict__ out) {
    extern __shared__ __bf16 Bs[];         // padded: byte offset = lds_off(elem)
    char* bsc = (char*)Bs;
    const int tid = threadIdx.x;
    const int n0 = blockIdx.x * 64;
    const int m0 = blockIdx.y * 128;
    __bf16* panel = Wb + (size_t)n0 * D_;  // 64 contiguous rows = whole panel

#if defined(STAGE_TDM)
    // One TDM op per block: 2D tile 768(k) x 64(n), 2B elements, LDS padding
    // 8 DWORDs every 128 DWORDs (pad_interval=6, pad_amount=7) -> lds_off().
    if (tid < 32) {
        unsigned long long ga = (unsigned long long)(uintptr_t)panel;
        v4u g0 = { 1u,                                   // count=1, user descriptor
                   0u,                                   // lds_addr = 0 (dyn-LDS base)
                   (unsigned int)(ga & 0xFFFFFFFFu),
                   (unsigned int)(((ga >> 32) & 0x1FFFFFFu) | 0x80000000u) }; // type=2
        v8i g1 = { (int)0x0F910000,   // data_size=2B, pad_en, pad_int=6, pad_amt=7
                   (int)0x03000000,   // tensor_dim0 = 768 (low 16 in [31:16])
                   (int)0x03000000,   // tensor_dim1 = 768
                   (int)0x03000000,   // tile_dim0   = 768
                   0x40,              // tile_dim1   = 64
                   D_,                // tensor_dim0_stride = 768 (low 32)
                   0, 0 };
        v4i gz = { 0, 0, 0, 0 };
#if __clang_major__ >= 23
        v8i gz8 = { 0, 0, 0, 0, 0, 0, 0, 0 };
        __builtin_amdgcn_tensor_load_to_lds(g0, g1, gz, gz, gz8, 0);
#else
        __builtin_amdgcn_tensor_load_to_lds(g0, g1, gz, gz, 0);
#endif
        __builtin_amdgcn_s_wait_tensorcnt(0);
    }
    __syncthreads();
#elif defined(STAGE_ASYNC)
    for (int i = tid; i < BPANEL_ELEMS / 8; i += 256) {
        int l = i * 8;
        __builtin_amdgcn_global_load_async_to_lds_b128(
            (__attribute__((address_space(1))) void*)(panel + l),
            (__attribute__((address_space(3))) void*)(bsc + lds_off(l)),
            0, 0);
    }
    __builtin_amdgcn_s_wait_asynccnt(0);
    __syncthreads();
#else
    for (int i = tid; i < BPANEL_ELEMS / 8; i += 256) {
        int l = i * 8;
        *(v8bf*)(bsc + lds_off(l)) = *(const v8bf*)(panel + l);
    }
    __syncthreads();
#endif

    const int wave = tid >> 5, lane = tid & 31;
    const int lo = lane & 15, hi = lane >> 4;
    const int mrow = m0 + wave * 16 + lo;
    const __bf16* Arow = Ab + (size_t)mrow * D_;

    // prefetch epilogue operands (x/xi tile) behind the WMMA K-loop
    {
        size_t pro = (size_t)mrow * D_ + n0 + hi * 32;
        __builtin_prefetch(&x[pro], 0, 1);
        __builtin_prefetch(&xi[pro], 0, 1);
    }

    // Loop-invariant B-fragment base pointers (row = j*16+lo, half = hi):
    //   lds_off((j*16+lo)*768 + hi*16 + kt) == ROWPITCH*(j*16+lo) + 32*hi
    //                                          + (2*kt + 32*(kt>>8))
    // (kt mod 256 <= 224, so the hi*16 term never carries into l>>8.)
    const char* bp[4];
    #pragma unroll
    for (int j = 0; j < 4; ++j)
        bp[j] = bsc + ROWPITCH * (j * 16 + lo) + 32 * hi;

    v8f acc[4] = {};
    #pragma unroll
    for (int kt = 0; kt < D_; kt += 32) {
        const int koff = 2 * kt + 32 * (kt >> 8);   // compile-time constant
        // A fragment (16x32 bf16): lane lo = M row; hi picks K-halves
        union { v16bf v; v8bf h[2]; } af;
        af.h[0] = *(const v8bf*)&Arow[kt + hi * 16];
        af.h[1] = *(const v8bf*)&Arow[kt + hi * 16 + 8];
        #pragma unroll
        for (int j = 0; j < 4; ++j) {
            // B fragment (32x16 bf16): N = lo, 16 consecutive K from kt + hi*16
            union { v16bf v; v8bf h[2]; } bf_;
            bf_.h[0] = *(const v8bf*)(bp[j] + koff);
            bf_.h[1] = *(const v8bf*)(bp[j] + koff + 16);
            acc[j] = __builtin_amdgcn_wmma_f32_16x16x32_bf16(
                false, af.v, false, bf_.v, (short)0, acc[j], false, false);
        }
    }

    // epilogue: exact f32 residual (g*x + gi*xi) + bias + GEMM
    float pbv[4];
    #pragma unroll
    for (int j = 0; j < 4; ++j) pbv[j] = projb[n0 + j * 16 + lo];
    #pragma unroll
    for (int r = 0; r < 8; ++r) {
        const int m = m0 + wave * 16 + hi * 8 + r;   // D layout: VGPR r -> M
        const float gm = g[m], gim = gi[m];
        const size_t ro = (size_t)m * D_;
        #pragma unroll
        for (int j = 0; j < 4; ++j) {
            const int n = n0 + j * 16 + lo;
            float res = gm * x[ro + n] + gim * xi[ro + n];
            out[ro + n] = res + pbv[j] + acc[j][r];
        }
    }
}

// ---------------------------------------------------------------------------
extern "C" void kernel_launch(void* const* d_in, const int* in_sizes, int n_in,
                              void* d_out, int out_size, void* d_ws, size_t ws_size,
                              hipStream_t stream) {
    const float* x    = (const float*)d_in[0];
    const float* xi   = (const float*)d_in[1];
    const int*   mask = (const int*)  d_in[2];
    // d_in[3] = len_t (compile-time 64)
    const float* pwW  = (const float*)d_in[4];
    const float* pwb  = (const float*)d_in[5];
    const float* pW   = (const float*)d_in[6];
    const float* pb   = (const float*)d_in[7];
    float* out = (float*)d_out;

    char* w = (char*)d_ws;
    size_t off = 0;
    __bf16* Wb    = (__bf16*)(w + off); off += (size_t)D_ * D_ * 2;        // 1.18 MB
    __bf16* Ab    = (__bf16*)(w + off); off += (size_t)MROWS * D_ * 2;     // 151 MB
    float*  Wproj = (float*) (w + off); off += (size_t)B_ * S_ * 4;
    float*  wv    = (float*) (w + off); off += (size_t)B_ * 4;
    float*  wiv   = (float*) (w + off); off += (size_t)B_ * 4;
    float*  g     = (float*) (w + off); off += (size_t)MROWS * 4;
    float*  gi    = (float*) (w + off); off += (size_t)MROWS * 4;

    k_cvt_bf16<<<(D_ * D_ + 255) / 256, 256, 0, stream>>>(pW, Wb, D_ * D_);
    k_stats<<<B_, 256, 0, stream>>>(x, xi, mask, pwW, pwb, Wproj, wv, wiv);
    k_build_a<<<MROWS, 256, 0, stream>>>(x, xi, mask, Wproj, wv, wiv, Ab, g, gi);
    k_gemm<<<dim3(D_ / 64, MROWS / 128), 256, BPANEL_BYTES, stream>>>(
        Ab, Wb, x, xi, g, gi, pb, out);
}
